// MultiHeadAttention_18640158064677
// MI455X (gfx1250) — matrix operand order repaired and checked by reading, not compiled, not run on hardware
//
#include <hip/hip_runtime.h>
#include <hip/hip_bf16.h>

// ---------------------------------------------------------------------------
// MultiHeadAttention forward for MI455X (gfx1250, wave32, WMMA).
//
// Shapes: B=1, S=4096, D=1024, H=16, DK=64.
// d_out = out[4096,1024] f32  ||  attn[16,4096,4096] f32.
//
// Roofline: ~137 GFLOP matrix math vs ~2.2 GB mandatory HBM traffic
// (attn write 1.07GB + mask read 1.07GB streamed twice -> NT hints so they
// bypass/evict-first in L2, keeping the 16 MB bf16 K/V resident in the
// 192 MB L2). bf16 WMMA (f32 accum) keeps the math at tens of microseconds
// so the kernel rides the 23.3 TB/s HBM roof. Scores are RECOMPUTED in
// pass 2 instead of spilled (34 GF of WMMA is far cheaper than 2.1 GB of
// score spill/reload traffic).
//
// vs round 1: software-pipelined K loops (no s_wait_loadcnt 0 right before
// each WMMA), per-lane online softmax stats (cross-lane merge hoisted out
// of the 256-iteration loop), NT cache hints, vectorized conversions.
// ---------------------------------------------------------------------------

#define SLEN   4096
#define DMODEL 1024
#define NHEAD  16
#define DKH    64

typedef unsigned short u16;
typedef unsigned int   u32;
typedef __attribute__((ext_vector_type(16))) __bf16 v16bf;
typedef __attribute__((ext_vector_type(8)))  float  v8f;
typedef __attribute__((ext_vector_type(4)))  float  f32x4;
typedef __attribute__((ext_vector_type(4)))  u16    u16x4;
typedef __attribute__((ext_vector_type(8)))  u16    u16x8;
typedef __attribute__((ext_vector_type(16))) u16    u16x16;

__device__ __forceinline__ u16 f2bf(float f) {          // round-to-nearest-even
  u32 u = __builtin_bit_cast(u32, f);
  u32 r = u + 0x7FFFu + ((u >> 16) & 1u);
  return (u16)(r >> 16);
}

__device__ __forceinline__ v16bf frag_from(u16x8 lo, u16x8 hi) {
  union { u16x16 w; u16x8 h[2]; } u;
  u.h[0] = lo; u.h[1] = hi;
  return __builtin_bit_cast(v16bf, u.w);
}

// A-matrix fragment (16x32, 16-bit). Per ISA table, lane half = lane>>4:
//   element i: K = half*8 + i          (i<8)
//              K = 16 + half*8 + (i-8) (i>=8)
__device__ __forceinline__ v16bf load_fragA(const u16* p, int half) {
  return frag_from(*(const u16x8*)(p + half * 8),
                   *(const u16x8*)(p + 16 + half * 8));
}

// B-matrix fragment (32x16, 16-bit). Same K order as A fragment:
//   element i: K = half*8 + i (i<8), K = 16 + half*8 + (i-8) (i>=8)
__device__ __forceinline__ v16bf load_fragB(const u16* p, int half) {
  return frag_from(*(const u16x8*)(p + half * 8),
                   *(const u16x8*)(p + 16 + half * 8));
}

__device__ __forceinline__ v8f wmma_bf16(v16bf a, v16bf b, v8f c) {
  // (neg_a, A, neg_b, B, c_mod, C, reuse_a, reuse_b)
  return __builtin_amdgcn_wmma_f32_16x16x32_bf16(false, a, false, b,
                                                 (short)0, c, false, false);
}

// ---------------------------------------------------------------------------
// f32 -> bf16 conversion, 4 elements/thread (b128 in, b64 out)
// ---------------------------------------------------------------------------
__global__ void cvt_f32_bf16(const float* __restrict__ x, u16* __restrict__ y,
                             int n4) {
  for (int i = blockIdx.x * blockDim.x + threadIdx.x; i < n4;
       i += gridDim.x * blockDim.x) {
    const f32x4 v = ((const f32x4*)x)[i];
    u16x4 o;
#pragma unroll
    for (int j = 0; j < 4; ++j) o[j] = f2bf(v[j]);
    ((u16x4*)y)[i] = o;
  }
}

// ---------------------------------------------------------------------------
// Y = X @ W^T + b.  X:[S,D] bf16, W:[D,D] bf16 row-major (W^T's column n is
// W's row n -> K-contiguous B fragments). Output bf16 (QKV) or f32 (out).
// One wave = 16x64 tile; double-buffered K loop.
// ---------------------------------------------------------------------------
__global__ __launch_bounds__(256) void gemm_xwt_bias(
    const u16* __restrict__ X, const u16* __restrict__ W,
    const float* __restrict__ bias, u16* __restrict__ Yb,
    float* __restrict__ Yf) {
  const int lane = threadIdx.x & 31;
  const int half = lane >> 4;
  const int n    = lane & 15;
  const int w    = blockIdx.x * 8 + (threadIdx.x >> 5);
  const int mt   = w >> 4;   // 256 M-tiles (16 rows)
  const int nt   = w & 15;   // 16 N-tiles (64 cols)

  const u16* xrow = X + (size_t)(mt * 16 + n) * DMODEL;
  const u16* wr[4];
#pragma unroll
  for (int j = 0; j < 4; ++j)
    wr[j] = W + (size_t)(nt * 64 + j * 16 + n) * DMODEL;

  // prologue: fragments for k0 = 0
  v16bf a = load_fragA(xrow, half);
  v16bf b[4];
#pragma unroll
  for (int j = 0; j < 4; ++j) b[j] = load_fragB(wr[j], half);

  v8f acc[4] = {};
#pragma unroll 2
  for (int k0 = 0; k0 < DMODEL; k0 += 32) {
    // prefetch next iteration (clamped on last iter; loads overlap WMMAs)
    const int kn = (k0 + 32 < DMODEL) ? k0 + 32 : k0;
    v16bf an = load_fragA(xrow + kn, half);
    v16bf bn[4];
#pragma unroll
    for (int j = 0; j < 4; ++j) bn[j] = load_fragB(wr[j] + kn, half);
#pragma unroll
    for (int j = 0; j < 4; ++j) acc[j] = wmma_bf16(a, b[j], acc[j]);
    a = an;
#pragma unroll
    for (int j = 0; j < 4; ++j) b[j] = bn[j];
  }

#pragma unroll
  for (int j = 0; j < 4; ++j) {
    const int col = nt * 64 + j * 16 + n;
    const float bv = bias[col];
#pragma unroll
    for (int r = 0; r < 8; ++r) {
      const int row = mt * 16 + r + 8 * half;   // C layout: c[r]=C[r+8*half][n]
      const float y = acc[j][r] + bv;
      if (Yb) Yb[(size_t)row * DMODEL + col] = f2bf(y);
      else    Yf[(size_t)row * DMODEL + col] = y;
    }
  }
}

// ---------------------------------------------------------------------------
// Vb [S,D] bf16 -> Vt [D,S] bf16 (per-head column-major V so P@V B-frags are
// K-contiguous). 32x32 LDS tile transpose.
// ---------------------------------------------------------------------------
__global__ void transpose_bf16(const u16* __restrict__ in,
                               u16* __restrict__ out) {
  __shared__ u16 tile[32][33];
  const int bx = blockIdx.x, by = blockIdx.y;
  const int tx = threadIdx.x, ty = threadIdx.y;
#pragma unroll
  for (int i = 0; i < 32; i += 8)
    tile[ty + i][tx] = in[(size_t)(by * 32 + ty + i) * DMODEL + bx * 32 + tx];
  __syncthreads();
#pragma unroll
  for (int i = 0; i < 32; i += 8)
    out[(size_t)(bx * 32 + ty + i) * SLEN + by * 32 + tx] = tile[tx][ty + i];
}

// ---------------------------------------------------------------------------
// Attention: one wave owns (head h, 16 query rows).
//   pass 1: WMMA QK^T tiles -> PER-LANE online (m,l); single cross-lane
//           merge after the loop (each lane sees one column per tile, the
//           16 lanes of a half jointly cover the row).
//   pass 2: recompute scores (all K/V fragment loads issued up front each
//           iteration), p = exp(s-m)/l, NT-write attn, stage p as bf16 via
//           per-wave LDS (C-frag -> A-frag re-layout), WMMA P@V.
// ---------------------------------------------------------------------------
__global__ __launch_bounds__(128) void attn_fused(
    const u16* __restrict__ Qb, const u16* __restrict__ Kb,
    const u16* __restrict__ Vt, const int* __restrict__ mask,
    float* __restrict__ attn, u16* __restrict__ ctxb) {
  __shared__ u16 Pbuf[4][16 * 32];   // per-wave P staging (1 KB each)

  const int wv   = threadIdx.x >> 5;
  const int lane = threadIdx.x & 31;
  const int half = lane >> 4;
  const int n    = lane & 15;
  const int gw   = blockIdx.x * 4 + wv;  // 0..4095
  const int h    = gw >> 8;
  const int qt   = gw & 255;
  const int qbase = qt * 16;

  // Q A-fragments: DK=64 -> two 16x32 fragments, resident for both passes
  const u16* qrow = Qb + (size_t)(qbase + n) * DMODEL + h * DKH;
  const v16bf aq0 = load_fragA(qrow,      half);
  const v16bf aq1 = load_fragA(qrow + 32, half);

  const float scale = 0.125f;   // 1/sqrt(64)

  // ---------------- pass 1: per-lane online softmax stats ---------------
  float ml[8], ll[8];
#pragma unroll
  for (int r = 0; r < 8; ++r) { ml[r] = -1e30f; ll[r] = 0.f; }

  const u16* kr0 = Kb + (size_t)n * DMODEL + h * DKH;
  v16bf b0 = load_fragB(kr0, half);
  v16bf b1 = load_fragB(kr0 + 32, half);
  for (int kb = 0; kb < SLEN / 16; ++kb) {
    const int kbn = (kb + 1 < SLEN / 16) ? kb + 1 : kb;   // clamped prefetch
    const u16* krn = Kb + (size_t)(kbn * 16 + n) * DMODEL + h * DKH;
    const v16bf nb0 = load_fragB(krn, half);
    const v16bf nb1 = load_fragB(krn + 32, half);

    v8f c = {};
    c = wmma_bf16(aq0, b0, c);
    c = wmma_bf16(aq1, b1, c);
#pragma unroll
    for (int r = 0; r < 8; ++r) {
      const int qr = qbase + r + 8 * half;
      const int mv =
          __builtin_nontemporal_load(&mask[(size_t)qr * SLEN + kb * 16 + n]);
      const float sv = c[r] * scale + (mv == 0 ? -1e9f : 0.f);
      const float nm = fmaxf(ml[r], sv);
      ll[r] = ll[r] * __expf(ml[r] - nm) + __expf(sv - nm);
      ml[r] = nm;
    }
    b0 = nb0; b1 = nb1;
  }

  // one-time cross-lane merge of (m,l) over the 16 lanes of each half
  float mrow[8], invl[8];
#pragma unroll
  for (int r = 0; r < 8; ++r) {
    float m = ml[r], l = ll[r];
#pragma unroll
    for (int x = 1; x < 16; x <<= 1) {
      const float mo = __shfl_xor(m, x);
      const float lo = __shfl_xor(l, x);
      const float nm = fmaxf(m, mo);
      l = l * __expf(m - nm) + lo * __expf(mo - nm);
      m = nm;
    }
    mrow[r] = m;
    invl[r] = 1.0f / l;
  }

  // ---------------- pass 2: write attn + ctx = P @ V --------------------
  v8f ctx[4] = {};
  u16* P = &Pbuf[wv][0];
  for (int kb2 = 0; kb2 < SLEN / 32; ++kb2) {
    // issue ALL fragment loads for this iteration first (K for both score
    // tiles + V for the P@V step) so WMMAs only wait on the oldest loads.
    const u16* kra = Kb + (size_t)(kb2 * 32 + n) * DMODEL + h * DKH;
    const u16* krb = Kb + (size_t)(kb2 * 32 + 16 + n) * DMODEL + h * DKH;
    const v16bf k0a = load_fragB(kra,      half);
    const v16bf k0b = load_fragB(kra + 32, half);
    const v16bf k1a = load_fragB(krb,      half);
    const v16bf k1b = load_fragB(krb + 32, half);
    v16bf vb[4];
#pragma unroll
    for (int j = 0; j < 4; ++j)
      vb[j] = load_fragB(Vt + (size_t)(h * DKH + j * 16 + n) * SLEN + kb2 * 32,
                         half);

    v8f c0 = {}, c1 = {};
    c0 = wmma_bf16(aq0, k0a, c0);
    c0 = wmma_bf16(aq1, k0b, c0);
    c1 = wmma_bf16(aq0, k1a, c1);
    c1 = wmma_bf16(aq1, k1b, c1);

#pragma unroll
    for (int sub = 0; sub < 2; ++sub) {
      const v8f c = sub ? c1 : c0;
#pragma unroll
      for (int r = 0; r < 8; ++r) {
        const int qr = qbase + r + 8 * half;
        const int kc = kb2 * 32 + sub * 16 + n;
        const int mv =
            __builtin_nontemporal_load(&mask[(size_t)qr * SLEN + kc]);
        const float p =
            __expf(c[r] * scale + (mv == 0 ? -1e9f : 0.f) - mrow[r]) * invl[r];
        __builtin_nontemporal_store(
            p, &attn[(size_t)h * SLEN * SLEN + (size_t)qr * SLEN + kc]);
        P[(r + 8 * half) * 32 + sub * 16 + n] = f2bf(p);  // A-layout staging
      }
    }
    // per-wave LDS RAW: DS ops are in-order within a wave; explicit wait for
    // clarity and to fence compiler reordering.
    asm volatile("s_wait_dscnt 0" ::: "memory");
    const v16bf pa = load_fragA(P + n * 32, half);        // P as 16x32 A-frag
#pragma unroll
    for (int j = 0; j < 4; ++j)
      ctx[j] = wmma_bf16(pa, vb[j], ctx[j]);
  }

  // ctx [16 x 64] -> bf16 into ctxb[S,D] at column block h*64
#pragma unroll
  for (int j = 0; j < 4; ++j)
#pragma unroll
    for (int r = 0; r < 8; ++r) {
      const int row = qbase + r + 8 * half;
      const int col = h * DKH + j * 16 + n;
      ctxb[(size_t)row * DMODEL + col] = f2bf(ctx[j][r]);
    }
}

// ---------------------------------------------------------------------------
extern "C" void kernel_launch(void* const* d_in, const int* in_sizes, int n_in,
                              void* d_out, int out_size, void* d_ws,
                              size_t ws_size, hipStream_t stream) {
  const float* query = (const float*)d_in[0];
  const float* key   = (const float*)d_in[1];
  const float* value = (const float*)d_in[2];
  const int*   mask  = (const int*)d_in[3];
  const float* Wq = (const float*)d_in[4];  const float* bq = (const float*)d_in[5];
  const float* Wk = (const float*)d_in[6];  const float* bk = (const float*)d_in[7];
  const float* Wv = (const float*)d_in[8];  const float* bv = (const float*)d_in[9];
  const float* Wo = (const float*)d_in[10]; const float* bo = (const float*)d_in[11];

  float* out  = (float*)d_out;
  float* attn = out + (size_t)SLEN * DMODEL;

  const size_t SD = (size_t)SLEN * DMODEL;   // 4M elems
  const size_t DD = (size_t)DMODEL * DMODEL; // 1M elems

  // Workspace (bf16 u16 elems), ~56 MB total with reuse:
  u16* ws  = (u16*)d_ws;
  u16* Xq  = ws;            // 8 MB  (reused as Vt after Q gemm)
  u16* Xk  = Xq + SD;       // 8 MB  (reused as Ctx after K gemm)
  u16* Xv  = Xk + SD;       // 8 MB
  u16* Wqb = Xv + SD;       // 2 MB
  u16* Wkb = Wqb + DD;      // 2 MB
  u16* Wvb = Wkb + DD;      // 2 MB
  u16* Wob = Wvb + DD;      // 2 MB
  u16* Qb  = Wob + DD;      // 8 MB
  u16* Kb  = Qb + SD;       // 8 MB
  u16* Vb  = Kb + SD;       // 8 MB
  u16* Vt  = Xq;            // reuse: Xq dead after Q projection
  u16* Ctx = Xk;            // reuse: Xk dead after K projection

  // 1) f32 -> bf16 conversions (4 elems/thread)
  cvt_f32_bf16<<<2048, 256, 0, stream>>>(query, Xq, (int)(SD / 4));
  cvt_f32_bf16<<<2048, 256, 0, stream>>>(key,   Xk, (int)(SD / 4));
  cvt_f32_bf16<<<2048, 256, 0, stream>>>(value, Xv, (int)(SD / 4));
  cvt_f32_bf16<<<512,  256, 0, stream>>>(Wq, Wqb, (int)(DD / 4));
  cvt_f32_bf16<<<512,  256, 0, stream>>>(Wk, Wkb, (int)(DD / 4));
  cvt_f32_bf16<<<512,  256, 0, stream>>>(Wv, Wvb, (int)(DD / 4));
  cvt_f32_bf16<<<512,  256, 0, stream>>>(Wo, Wob, (int)(DD / 4));

  // 2) QKV projections (bf16 WMMA, bf16 outputs)
  gemm_xwt_bias<<<512, 256, 0, stream>>>(Xq, Wqb, bq, Qb, nullptr);
  gemm_xwt_bias<<<512, 256, 0, stream>>>(Xk, Wkb, bk, Kb, nullptr);
  gemm_xwt_bias<<<512, 256, 0, stream>>>(Xv, Wvb, bv, Vb, nullptr);

  // 3) V -> per-head column-major for K-contiguous P@V B-fragments
  transpose_bf16<<<dim3(DMODEL / 32, SLEN / 32), dim3(32, 8), 0, stream>>>(Vb, Vt);

  // 4) fused flash-style attention: writes attn (f32, NT) + ctx (bf16)
  attn_fused<<<1024, 128, 0, stream>>>(Qb, Kb, Vt, mask, attn, Ctx);

  // 5) output projection (f32 output into d_out)
  gemm_xwt_bias<<<512, 256, 0, stream>>>(Ctx, Wob, bo, nullptr, out);
}
